// FlaxGPTNeoSelfAttention_4337916969011
// MI455X (gfx1250) — matrix operand order repaired and checked by reading, not compiled, not run on hardware
//
#include <hip/hip_runtime.h>

// ---------------------------------------------------------------------------
// GPT-Neo local self-attention for gfx1250 (MI455X), wave32 + WMMA bf16.
//   B=2, S=2048, D=2048, H=16, HD=128, WINDOW=256
// Pipeline: f32->bf16 convert / weight transpose, 3x WMMA GEMM (Q,K,V) with
// double-buffered TDM (tensor_load_to_lds) tile staging, flash-style windowed
// attention (WMMA for QK^T and P*V, async-to-LDS K staging), WMMA GEMM + bias.
// ---------------------------------------------------------------------------

typedef __attribute__((ext_vector_type(16))) __bf16 v16bf;
typedef __attribute__((ext_vector_type(8)))  float  v8f;
typedef __attribute__((ext_vector_type(4)))  __bf16 v4bf;

union AFrag { v16bf v; uint4 q[2]; };

#define WMMA_BF16(a, b, c) \
  __builtin_amdgcn_wmma_f32_16x16x32_bf16(false, (a), false, (b), (short)0, (c), false, false)

static constexpr int kB = 2, kS = 2048, kD = 2048, kH = 16, kHD = 128, kWIN = 256;
static constexpr int kM = kB * kS;   // 4096 rows in all GEMMs

#if __has_builtin(__builtin_amdgcn_tensor_load_to_lds) && \
    __has_builtin(__builtin_amdgcn_s_wait_tensorcnt)
#define USE_TDM 1
#else
#define USE_TDM 0
#endif

// Low 32 bits of a generic pointer to LDS == LDS byte offset (aperture form).
__device__ inline uint32_t lds_off(const void* p) {
  return (uint32_t)(uintptr_t)p;
}

// GLOBAL_LOAD_ASYNC_TO_LDS_B128: per-lane 16B DMA global -> LDS, ASYNCcnt.
__device__ inline void async_ld_b128(uint32_t lds, const void* g) {
  asm volatile("global_load_async_to_lds_b128 %0, %1, off"
               :: "v"(lds), "v"((unsigned long long)g) : "memory");
}

#if USE_TDM
// Tensor Data Mover: DMA a 2D tile (tileX x tileY elements, 2B/elem) from a
// row-major tensor (row stride = strideElems) into LDS, with LDS padding of
// padAmt+1 DWORDs inserted every 2^(padIv+1) DWORDs (hardware row padding).
__device__ inline void tdm_load_tile_2d(uint32_t ldsAddr, unsigned long long ga,
                                        uint32_t tileX, uint32_t tileY,
                                        uint32_t strideElems,
                                        uint32_t padIv, uint32_t padAmt) {
  typedef unsigned int v4u_t __attribute__((ext_vector_type(4)));
  typedef int v8i_t __attribute__((ext_vector_type(8)));
  typedef int v4i_t __attribute__((ext_vector_type(4)));
  const uint32_t dim0 = 0x100000u, dim1 = 0x100000u;   // generous (no OOB clip)
  v4u_t g0;
  g0[0] = 1u;                                      // count=1, user descriptor
  g0[1] = ldsAddr;                                 // lds_addr
  g0[2] = (uint32_t)ga;                            // global_addr[31:0]
  g0[3] = (uint32_t)(ga >> 32) | 0x80000000u;      // global_addr[56:32] | type=2
  v8i_t g1;
  g1[0] = (int)((1u << 16) | (1u << 20) | (padIv << 22) | (padAmt << 25));
  g1[1] = (int)((dim0 & 0xFFFFu) << 16);                           // tensor_dim0 lo
  g1[2] = (int)(((dim0 >> 16) & 0xFFFFu) | ((dim1 & 0xFFFFu) << 16));
  g1[3] = (int)(((dim1 >> 16) & 0xFFFFu) | (tileX << 16));         // tile_dim0
  g1[4] = (int)(tileY & 0xFFFFu);                                  // tile_dim1
  g1[5] = (int)strideElems;                                        // dim0_stride lo
  g1[6] = 0;
  g1[7] = 0;
  v4i_t z4 = (v4i_t)0;                              // 2D tensor: groups 2/3 unused
  v8i_t z8 = (v8i_t)0;
  __builtin_amdgcn_tensor_load_to_lds(g0, g1, z4, z4, z8, 0);
}
#endif

// ---------------------------------------------------------------------------
// f32 -> bf16 elementwise convert (4 elements/thread, vectorized)
// ---------------------------------------------------------------------------
__global__ __launch_bounds__(256)
void cvt_f32_bf16_x4(const float4* __restrict__ in, v4bf* __restrict__ out, int n4) {
  int i = blockIdx.x * 256 + threadIdx.x;
  if (i < n4) {
    float4 v = in[i];
    v4bf o;
    o[0] = (__bf16)v.x; o[1] = (__bf16)v.y; o[2] = (__bf16)v.z; o[3] = (__bf16)v.w;
    out[i] = o;
  }
}

// ---------------------------------------------------------------------------
// W[K][N] f32  ->  Wt[N][K] bf16   (tiled 32x32 transpose through LDS)
// ---------------------------------------------------------------------------
__global__ __launch_bounds__(256)
void conv_transpose_w(const float* __restrict__ W, __bf16* __restrict__ Wt, int K, int N) {
  __shared__ float tile[32][33];
  int bx = blockIdx.x * 32;          // N offset
  int by = blockIdx.y * 32;          // K offset
  int tx = threadIdx.x & 31;
  int ty = threadIdx.x >> 5;         // 0..7
#pragma unroll
  for (int i = 0; i < 32; i += 8)
    tile[ty + i][tx] = W[(size_t)(by + ty + i) * N + bx + tx];
  __syncthreads();
#pragma unroll
  for (int i = 0; i < 32; i += 8)
    Wt[(size_t)(bx + ty + i) * K + by + tx] = (__bf16)tile[tx][ty + i];
}

// ---------------------------------------------------------------------------
// WMMA GEMM: C[M][N] = A[M][K] * Bt[N][K]^T (+bias).  bf16 in, f32 accum.
// Block tile 128x128, 8 waves (2x4), wave tile 64x32 -> 4x2 wmma tiles.
// Double-buffered LDS staged by the Tensor Data Mover (one wave issues two
// descriptors per k-tile; TENSORcnt in-order semantics keep the next tile in
// flight while the current one feeds the WMMA stream). Async-LDS fallback.
// ---------------------------------------------------------------------------
template <bool F32OUT>
__global__ __launch_bounds__(256)
void gemm_wmma_bf16(const __bf16* __restrict__ A, const __bf16* __restrict__ Bt,
                    const float* __restrict__ bias,
                    __bf16* __restrict__ outB, float* __restrict__ outF,
                    int M, int N, int K) {
  __shared__ __align__(16) __bf16 As[2][128][40];   // 64B row + 16B pad (TDM pad)
  __shared__ __align__(16) __bf16 Bs[2][128][40];

  const int tid   = threadIdx.x;
  const int lane  = tid & 31;
  const int wid   = tid >> 5;
  const int waveM = wid >> 2;              // 0..1  (64-row stripe)
  const int waveN = wid & 3;               // 0..3  (32-col stripe)
  const int hlf   = lane >> 4;             // lane half (K-striping select)
  const int l16   = lane & 15;
  const int mBlk  = blockIdx.y * 128;
  const int nBlk  = blockIdx.x * 128;

#if USE_TDM
  const unsigned long long gAbase = (unsigned long long)(uintptr_t)(A  + (size_t)mBlk * K);
  const unsigned long long gBbase = (unsigned long long)(uintptr_t)(Bt + (size_t)nBlk * K);
  const uint32_t ldsA[2] = { lds_off(&As[0][0][0]), lds_off(&As[1][0][0]) };
  const uint32_t ldsB[2] = { lds_off(&Bs[0][0][0]), lds_off(&Bs[1][0][0]) };
  const bool issuer = (wid == 0);
  auto issue = [&](int bf, int it) {
    // k-offset in bytes: it * 32 elems * 2B. padIv=3 -> 64B interval, padAmt=3 -> 16B.
    tdm_load_tile_2d(ldsA[bf], gAbase + (unsigned long long)it * 64u, 32u, 128u,
                     (uint32_t)K, 3u, 3u);
    tdm_load_tile_2d(ldsB[bf], gBbase + (unsigned long long)it * 64u, 32u, 128u,
                     (uint32_t)K, 3u, 3u);
  };
#else
  // Per-thread async staging slots: two 16B chunks per tile per matrix.
  const int r0 = tid >> 2, s0 = (tid & 3) * 8;
  const int r1 = r0 + 64;
  const __bf16* gA0 = A  + (size_t)(mBlk + r0) * K + s0;
  const __bf16* gA1 = A  + (size_t)(mBlk + r1) * K + s0;
  const __bf16* gB0 = Bt + (size_t)(nBlk + r0) * K + s0;
  const __bf16* gB1 = Bt + (size_t)(nBlk + r1) * K + s0;
  uint32_t lA0[2], lA1[2], lB0[2], lB1[2];
#pragma unroll
  for (int bf = 0; bf < 2; ++bf) {
    lA0[bf] = lds_off(&As[bf][r0][s0]);
    lA1[bf] = lds_off(&As[bf][r1][s0]);
    lB0[bf] = lds_off(&Bs[bf][r0][s0]);
    lB1[bf] = lds_off(&Bs[bf][r1][s0]);
  }
  auto issue = [&](int bf, int it) {
    int kk = it * 32;
    async_ld_b128(lA0[bf], gA0 + kk);
    async_ld_b128(lA1[bf], gA1 + kk);
    async_ld_b128(lB0[bf], gB0 + kk);
    async_ld_b128(lB1[bf], gB1 + kk);
  };
#endif

  v8f acc[4][2] = {};

  const int nIter = K / 32;
#if USE_TDM
  if (issuer) issue(0, 0);
#else
  issue(0, 0);
#endif
  for (int it = 0; it < nIter; ++it) {
    const int cur = it & 1;
    const bool more = (it + 1) < nIter;
#if USE_TDM
    if (issuer) {
      if (more) {
        issue(1 - cur, it + 1);                    // prefetch next tile
        __builtin_amdgcn_s_wait_tensorcnt(2);      // cur's 2 descriptors landed
      } else {
        __builtin_amdgcn_s_wait_tensorcnt(0);
      }
    }
#else
    if (more) issue(1 - cur, it + 1);
    if (more) asm volatile("s_wait_asynccnt 0x4" ::: "memory");
    else      asm volatile("s_wait_asynccnt 0x0" ::: "memory");
#endif
    __syncthreads();

    AFrag af[4], bfm[2];
#pragma unroll
    for (int i = 0; i < 4; ++i) {
      int row = waveM * 64 + i * 16 + l16;
      af[i].q[0] = *(const uint4*)&As[cur][row][hlf * 16];       // K = hlf*16 + 0..7
      af[i].q[1] = *(const uint4*)&As[cur][row][hlf * 16 + 8];   // K = hlf*16 + 8..15
    }
#pragma unroll
    for (int j = 0; j < 2; ++j) {
      int col = waveN * 32 + j * 16 + l16;
      bfm[j].q[0] = *(const uint4*)&Bs[cur][col][hlf * 16];      // K = hlf*16 + 0..7
      bfm[j].q[1] = *(const uint4*)&Bs[cur][col][hlf * 16 + 8];  // K = hlf*16 + 8..15
    }
#pragma unroll
    for (int i = 0; i < 4; ++i)
#pragma unroll
      for (int j = 0; j < 2; ++j)
        acc[i][j] = WMMA_BF16(af[i].v, bfm[j].v, acc[i][j]);
    __syncthreads();                         // all reads of cur done before reuse
  }

  // Epilogue. C layout: VGPR r, lane -> M = r + 8*(lane/16), N = lane%16.
#pragma unroll
  for (int i = 0; i < 4; ++i) {
#pragma unroll
    for (int j = 0; j < 2; ++j) {
      int col = nBlk + waveN * 32 + j * 16 + l16;
      float bv = F32OUT ? bias[col] : 0.0f;
#pragma unroll
      for (int r = 0; r < 8; ++r) {
        int row = mBlk + waveM * 64 + i * 16 + r + 8 * hlf;
        float v = acc[i][j][r];
        if (F32OUT) outF[(size_t)row * N + col] = v + bv;
        else        outB[(size_t)row * N + col] = (__bf16)v;
      }
    }
  }
}

// ---------------------------------------------------------------------------
// Windowed flash attention. One block = (b, h, 64 query rows).
// 8 waves: 4 q-stripes (16 rows) x 2 hd-halves (64 cols).
// Key strip [qStart-256, qStart+63] processed in 10 chunks of 32 keys.
// K chunks stream async into LDS while the VALU transposes V.
// ---------------------------------------------------------------------------
__global__ __launch_bounds__(256)
void attn_local(const __bf16* __restrict__ Q, const __bf16* __restrict__ Kg,
                const __bf16* __restrict__ Vg, const int* __restrict__ amask,
                __bf16* __restrict__ O) {
  __shared__ __align__(16) __bf16 Qs[64][136];   // q rows x HD (padded)
  __shared__ __align__(16) __bf16 Ks[32][136];   // key rows x HD
  __shared__ __align__(16) __bf16 Vt[128][40];   // HD x 32 keys (transposed)
  __shared__ __align__(16) __bf16 Ps[8][16][40]; // per-wave P relayout buffer

  const int qb = blockIdx.x, h = blockIdx.y, b = blockIdx.z;
  const int qStart = qb * 64;
  const int tid  = threadIdx.x;
  const int lane = tid & 31, wid = tid >> 5;
  const int stripe = wid & 3;            // q-stripe (16 rows)
  const int nHalf  = wid >> 2;           // hd half (64 cols)
  const int hlf = lane >> 4, l16 = lane & 15;

  // Load Q tile 64x128 bf16 (1024 x 16B chunks, 4 per thread).
#pragma unroll
  for (int i = 0; i < 4; ++i) {
    int c = tid + i * 256;
    int row = c >> 4, seg = c & 15;
    size_t g = ((size_t)((b * kS + qStart + row) * kH + h)) * kHD + seg * 8;
    *(uint4*)&Qs[row][seg * 8] = *(const uint4*)(Q + g);
  }
  __syncthreads();

  // Cache this wave's Q A-fragments for the 4 HD k-steps.
  AFrag qf[4];
  {
    int row = stripe * 16 + l16;
#pragma unroll
    for (int s = 0; s < 4; ++s) {
      qf[s].q[0] = *(const uint4*)&Qs[row][s * 32 + hlf * 16];
      qf[s].q[1] = *(const uint4*)&Qs[row][s * 32 + hlf * 16 + 8];
    }
  }

  v8f acc[4] = {};
  float mRun[8], lSum[8];
#pragma unroll
  for (int r = 0; r < 8; ++r) { mRun[r] = -1e30f; lSum[r] = 0.0f; }

  const int sBase = qStart - kWIN;       // may be negative; masked out
  for (int ch = 0; ch < 10; ++ch) {
    const int k0 = sBase + ch * 32;
    __syncthreads();                     // protect Ks/Vt reuse
    // K chunk: async global->LDS (ASYNCcnt).  V chunk: load + transpose scatter.
#pragma unroll
    for (int i = 0; i < 2; ++i) {
      int c = tid + i * 256;             // 0..511
      int row = c >> 4, seg = c & 15;
      int j = k0 + row; int jc = j < 0 ? 0 : j;
      size_t g = ((size_t)((b * kS + jc) * kH + h)) * kHD + seg * 8;
      async_ld_b128(lds_off(&Ks[row][seg * 8]), Kg + g);
      uint4 vv = *(const uint4*)(Vg + g);
      const __bf16* ve = (const __bf16*)&vv;
#pragma unroll
      for (int e = 0; e < 8; ++e) Vt[seg * 8 + e][row] = ve[e];
    }
    asm volatile("s_wait_asynccnt 0x0" ::: "memory");
    __syncthreads();

    // QK^T: scores for 16 q rows x 32 keys (two 16x16 C tiles).
    v8f s0 = {}, s1 = {};
#pragma unroll
    for (int st = 0; st < 4; ++st) {
      AFrag kb0, kb1;
      kb0.q[0] = *(const uint4*)&Ks[l16][st * 32 + hlf * 16];
      kb0.q[1] = *(const uint4*)&Ks[l16][st * 32 + hlf * 16 + 8];
      kb1.q[0] = *(const uint4*)&Ks[16 + l16][st * 32 + hlf * 16];
      kb1.q[1] = *(const uint4*)&Ks[16 + l16][st * 32 + hlf * 16 + 8];
      s0 = WMMA_BF16(qf[st].v, kb0.v, s0);
      s1 = WMMA_BF16(qf[st].v, kb1.v, s1);
    }

    const int jcol0 = k0 + l16;
    const int jcol1 = jcol0 + 16;
    const int am0 = (jcol0 >= 0) ? amask[b * kS + jcol0] : 0;
    const int am1 = (jcol1 >= 0) ? amask[b * kS + jcol1] : 0;

    // Online softmax per row (rows live in 16-lane halves, VGPR index r).
    float alphaV[8];
#pragma unroll
    for (int r = 0; r < 8; ++r) {
      int qi = qStart + stripe * 16 + r + 8 * hlf;
      float v0 = (am0 && jcol0 <= qi && jcol0 > qi - kWIN) ? s0[r] : -1e30f;
      float v1 = (am1 && jcol1 <= qi && jcol1 > qi - kWIN) ? s1[r] : -1e30f;
      float mloc = fmaxf(v0, v1);
#pragma unroll
      for (int off = 8; off; off >>= 1) mloc = fmaxf(mloc, __shfl_xor(mloc, off));
      float mNew = fmaxf(mRun[r], mloc);
      float p0 = __expf(v0 - mNew);
      float p1 = __expf(v1 - mNew);
      float rs = p0 + p1;
#pragma unroll
      for (int off = 8; off; off >>= 1) rs += __shfl_xor(rs, off);
      float alpha = __expf(mRun[r] - mNew);
      lSum[r] = lSum[r] * alpha + rs;
      mRun[r] = mNew;
      alphaV[r] = alpha;
      int prow = r + 8 * hlf;
      Ps[wid][prow][l16]      = (__bf16)p0;
      Ps[wid][prow][16 + l16] = (__bf16)p1;
    }
    // Wave-internal LDS store->load ordering for the P relayout.
    asm volatile("s_wait_dscnt 0" ::: "memory");

    // Rescale accumulators by per-row alpha.
#pragma unroll
    for (int t = 0; t < 4; ++t)
#pragma unroll
      for (int r = 0; r < 8; ++r) acc[t][r] *= alphaV[r];

    // P * V: A = P (16q x 32key), B = V (32key x 16hd) per tile.
    AFrag pf;
    pf.q[0] = *(const uint4*)&Ps[wid][l16][hlf * 16];
    pf.q[1] = *(const uint4*)&Ps[wid][l16][hlf * 16 + 8];
#pragma unroll
    for (int t = 0; t < 4; ++t) {
      AFrag vb;
      int col = nHalf * 64 + t * 16 + l16;
      vb.q[0] = *(const uint4*)&Vt[col][hlf * 16];
      vb.q[1] = *(const uint4*)&Vt[col][hlf * 16 + 8];
      acc[t] = WMMA_BF16(pf.v, vb.v, acc[t]);
    }
  }

  // Normalize and store bf16 attention output [B,S,H,HD].
#pragma unroll
  for (int r = 0; r < 8; ++r) {
    float inv = 1.0f / lSum[r];
    int qi = qStart + stripe * 16 + r + 8 * hlf;
#pragma unroll
    for (int t = 0; t < 4; ++t) {
      int col = nHalf * 64 + t * 16 + l16;
      size_t g = ((size_t)((b * kS + qi) * kH + h)) * kHD + col;
      O[g] = (__bf16)(acc[t][r] * inv);
    }
  }
}

// ---------------------------------------------------------------------------
// Host-side launch
// ---------------------------------------------------------------------------
extern "C" void kernel_launch(void* const* d_in, const int* in_sizes, int n_in,
                              void* d_out, int out_size, void* d_ws, size_t ws_size,
                              hipStream_t stream) {
  const float* hs = (const float*)d_in[0];
  const int*   am = (const int*)d_in[1];
  const float* Wq = (const float*)d_in[2];
  const float* Wk = (const float*)d_in[3];
  const float* Wv = (const float*)d_in[4];
  const float* Wo = (const float*)d_in[5];
  const float* bo = (const float*)d_in[6];
  float* out = (float*)d_out;

  const size_t BSD = (size_t)kB * kS * kD;   // 8,388,608
  const size_t DD  = (size_t)kD * kD;        // 4,194,304

  char* ws = (char*)d_ws;
  size_t off = 0;
  auto alloc = [&](size_t bytes) {
    void* p = ws + off;
    off += (bytes + 255) & ~(size_t)255;
    return p;
  };
  __bf16* Xb  = (__bf16*)alloc(BSD * 2);
  __bf16* Wqt = (__bf16*)alloc(DD * 2);
  __bf16* Wkt = (__bf16*)alloc(DD * 2);
  __bf16* Wvt = (__bf16*)alloc(DD * 2);
  __bf16* Wot = (__bf16*)alloc(DD * 2);
  __bf16* Qb  = (__bf16*)alloc(BSD * 2);
  __bf16* Kb  = (__bf16*)alloc(BSD * 2);
  __bf16* Vb  = (__bf16*)alloc(BSD * 2);
  __bf16* AOb = (__bf16*)alloc(BSD * 2);

  // 1) Convert activations to bf16.
  {
    int n4 = (int)(BSD / 4);
    cvt_f32_bf16_x4<<<(n4 + 255) / 256, 256, 0, stream>>>((const float4*)hs, (v4bf*)Xb, n4);
  }
  // 2) Convert + transpose weights to N-major bf16.
  dim3 tg(kD / 32, kD / 32);
  conv_transpose_w<<<tg, 256, 0, stream>>>(Wq, Wqt, kD, kD);
  conv_transpose_w<<<tg, 256, 0, stream>>>(Wk, Wkt, kD, kD);
  conv_transpose_w<<<tg, 256, 0, stream>>>(Wv, Wvt, kD, kD);
  conv_transpose_w<<<tg, 256, 0, stream>>>(Wo, Wot, kD, kD);

  // 3) Q/K/V projections (bf16 out).
  dim3 gg(kD / 128, kM / 128);   // (16, 32)
  gemm_wmma_bf16<false><<<gg, 256, 0, stream>>>(Xb, Wqt, nullptr, Qb, nullptr, kM, kD, kD);
  gemm_wmma_bf16<false><<<gg, 256, 0, stream>>>(Xb, Wkt, nullptr, Kb, nullptr, kM, kD, kD);
  gemm_wmma_bf16<false><<<gg, 256, 0, stream>>>(Xb, Wvt, nullptr, Vb, nullptr, kM, kD, kD);

  // 4) Windowed attention.
  attn_local<<<dim3(kS / 64, kH, kB), 256, 0, stream>>>(Qb, Kb, Vb, am, AOb);

  // 5) Output projection + bias (f32 out).
  gemm_wmma_bf16<true><<<gg, 256, 0, stream>>>(AOb, Wot, bo, nullptr, out, kM, kD, kD);
}
